// VirtualKinematicChain_8658654069228
// MI455X (gfx1250) — matrix operand order repaired and checked by reading, not compiled, not run on hardware
//
#include <hip/hip_runtime.h>

// ---------------------------------------------------------------------------
// VirtualKinematicChain — persistent bf16-WMMA kernel for MI455X (gfx1250)
// One workgroup = 16 batch rows, resident for all 64 timesteps.
// Weights are pre-shuffled ("fragmentized") into wave32 WMMA B-fragment
// layout so the hot loop issues only b128 loads.
// ---------------------------------------------------------------------------

typedef __bf16 bf16_t;
typedef __attribute__((ext_vector_type(16))) __bf16 v16bf;
typedef __attribute__((ext_vector_type(8)))  __bf16 v8bf;
typedef __attribute__((ext_vector_type(8)))  float  v8f;

#define B_TOTAL   1024
#define TSTEPS    64
#define HID       128
#define NJ        3
#define HEADS     4
#define DH        32
#define LAYERS    3
#define TILE_B    16
#define NBLOCKS   (B_TOTAL / TILE_B)   // 64
#define NTHREADS  256
#define NWAVES    8
#define NEG_SLOPE 0.2f
#define PHYS_W    0.1f

// workspace offsets (bf16 elements; all multiples of 512 -> 32B aligned)
#define OFF_JIW   0            // 3*256*128
#define OFF_GATW  98304        // 3*128*128
#define OFF_WIH   147456       // 128*512
#define OFF_WHH   212992       // 128*512
#define OFF_PW1   278528       // 128*64
#define OFF_AW1   286720       // 128*64
#define OFF_AW2   294912       // 64*32

// ---------------------------------------------------------------------------
// WMMA fragment helpers (layouts per CDNA5 ISA §7.12.2, wave32)
// ---------------------------------------------------------------------------

// A matrix 16x32 bf16, row-major src, leading dim ld (multiple of 8).
// lanes 0-15: rows 0-15, K {0..7, 16..23}; lanes 16-31: K {8..15, 24..31}.
// Two contiguous 16B runs per lane -> 2x ds_load_b128.
__device__ __forceinline__ v16bf load_frag_A(const bf16_t* src, int ld, int lane) {
  const int row   = lane & 15;
  const int kbase = (lane & 16) ? 8 : 0;
  const bf16_t* p = src + row * ld + kbase;
  v8bf lo = *(const v8bf*)p;
  v8bf hi = *(const v8bf*)(p + 16);
  return __builtin_shufflevector(lo, hi, 0, 1, 2, 3, 4, 5, 6, 7,
                                 8, 9, 10, 11, 12, 13, 14, 15);
}

// B fragment from pre-shuffled weights: tile = (nt * nKt + kt); each lane
// reads its 16 bf16 contiguously (32B) -> 2x global_load_b128.
__device__ __forceinline__ v16bf load_frag_Bp(const bf16_t* base, int tile, int lane) {
  return *(const v16bf*)(base + tile * 512 + lane * 16);
}

__device__ __forceinline__ v8f wmma_bf16(v16bf a, v16bf b, v8f c) {
  return __builtin_amdgcn_wmma_f32_16x16x32_bf16(false, a, false, b,
                                                 (short)0, c, false, false);
}

__device__ __forceinline__ float sigm(float x) { return 1.0f / (1.0f + __expf(-x)); }

// cheap hash-based ~N(0,1) surrogate (JAX threefry not reproducible here)
__device__ __forceinline__ float hash_noise(unsigned idx, unsigned t) {
  unsigned x = idx * 2654435761u ^ (t * 40503u + 0x9E3779B9u);
  x ^= x >> 16; x *= 0x7feb352du; x ^= x >> 15; x *= 0x846ca68bu; x ^= x >> 16;
  float u1 = (float)(x & 0xffffu) * (1.0f / 65536.0f);
  float u2 = (float)(x >> 16)     * (1.0f / 65536.0f);
  return (u1 + u2 - 1.0f) * 2.44949f; // var ~= 1
}

// ---------------------------------------------------------------------------
// weight repack: f32 row-major [slices][K][N] -> bf16 B-fragment layout.
// dst element index o (within slice): tile = o/512, lane = (o%512)/16,
// e = o%16 ; k = kt*32 + ((lane&16)?8:0) + (e&7) + ((e&8)?16:0) ; n = nt*16 + (lane&15)
// with nKt = K/32, nt = tile / nKt, kt = tile % nKt.
// ---------------------------------------------------------------------------
__global__ void repack_b_frag(const float* __restrict__ src,
                              bf16_t* __restrict__ dst,
                              int K, int N, int slices) {
  int i = blockIdx.x * blockDim.x + threadIdx.x;
  int per = K * N;
  if (i >= per * slices) return;
  int s = i / per, o = i % per;
  int tile = o >> 9, r = o & 511;
  int lane = r >> 4, e = r & 15;
  int nKt = K >> 5;
  int nt = tile / nKt, kt = tile % nKt;
  int k = kt * 32 + ((lane & 16) ? 8 : 0) + (e & 7) + ((e & 8) ? 16 : 0);
  int n = nt * 16 + (lane & 15);
  dst[i] = (bf16_t)src[s * per + k * N + n];
}

// ---------------------------------------------------------------------------
// main persistent kernel
// ---------------------------------------------------------------------------
__global__ __launch_bounds__(NTHREADS, 1)
void vkc_main(const int* __restrict__ gesture_idx,
              const float* __restrict__ demographics,
              const float* __restrict__ emb_table,
              const float* __restrict__ dW1, const float* __restrict__ db1,
              const float* __restrict__ dW2, const float* __restrict__ db2,
              const bf16_t* __restrict__ jiW, const float* __restrict__ jib,
              const bf16_t* __restrict__ gatW,
              const float* __restrict__ gat_asrc, const float* __restrict__ gat_adst,
              const float* __restrict__ gat_b,
              const float* __restrict__ ln_g, const float* __restrict__ ln_b,
              const bf16_t* __restrict__ W_ih, const bf16_t* __restrict__ W_hh,
              const float* __restrict__ b_ih, const float* __restrict__ b_hh,
              const bf16_t* __restrict__ pW1, const float* __restrict__ pb1,
              const float* __restrict__ pW2, const float* __restrict__ pb2,
              const bf16_t* __restrict__ aW1, const float* __restrict__ ab1,
              const bf16_t* __restrict__ aW2, const float* __restrict__ ab2,
              const float* __restrict__ aW3, const float* __restrict__ ab3,
              float* __restrict__ out) {
  const int tid  = threadIdx.x;
  const int lane = tid & 31;
  const int wave = tid >> 5;
  const int gb   = blockIdx.x * TILE_B;

  // persistent LDS (~48 KB with phase-lifetime aliasing)
  __shared__ __align__(32) bf16_t sX [NJ * TILE_B * HID];   // 12 KB [j][b][f]
  __shared__ __align__(32) bf16_t sXP[NJ * TILE_B * HID];   // 12 KB xp / head temps
  __shared__ __align__(32) bf16_t sG [TILE_B * 4 * HID];    // 16 KB gates / agg / init
  __shared__ __align__(32) bf16_t sH [TILE_B * HID];        //  4 KB LSTM hidden
  __shared__ float  sS [TILE_B * NJ * HEADS];
  __shared__ float  sD [TILE_B * NJ * HEADS];
  __shared__ float  sAl[TILE_B * NJ * NJ * HEADS];

  float*  sHid  = (float*)sG;          // init: demo hidden [16][64] f32 (4 KB)
  bf16_t* sComb = sG + 4096;           // init: combined [16][256] bf16 (8 KB)
  float*  sDemo = (float*)sXP;         // init: demo out  [16][128] f32 (8 KB)
  bf16_t* sAgg  = sG;                  // GAT: aggregated rows [48][128] bf16
  bf16_t* sP1   = sXP;                 // heads: [16][64]
  bf16_t* sA1   = sXP + 1024;          // heads: [16][64]
  bf16_t* sA2   = sXP + 2048;          // heads: [16][32]

  const int adj[NJ][NJ] = {{1,0,0},{1,1,0},{0,1,1}};

  // ---------------- init: demographics MLP + embedding + joint projection ---
  for (int p = tid; p < TILE_B * 64; p += NTHREADS) {
    int b = p >> 6, u = p & 63;
    float acc = db1[u];
#pragma unroll
    for (int k = 0; k < 7; ++k)
      acc += demographics[(gb + b) * 7 + k] * dW1[k * 64 + u];
    sHid[p] = fmaxf(acc, 0.0f);
  }
  __syncthreads();
  for (int p = tid; p < TILE_B * HID; p += NTHREADS) {
    int b = p >> 7, u = p & 127;
    float acc = db2[u];
    for (int k = 0; k < 64; ++k) acc += sHid[b * 64 + k] * dW2[k * HID + u];
    sDemo[p] = acc;
  }
  __syncthreads();
  for (int p = tid; p < TILE_B * 256; p += NTHREADS) {
    int b = p >> 8, f = p & 255;
    float v = (f < HID) ? emb_table[gesture_idx[gb + b] * HID + f]
                        : sDemo[b * HID + (f - HID)];
    sComb[p] = (bf16_t)v;
  }
  __syncthreads();
  // jf = relu(comb[16,256] @ jiW[j] + jib[j])  — WMMA, K=256 (nKt=8)
  for (int ti = wave; ti < NJ * 8; ti += NWAVES) {
    int j = ti >> 3, nt = ti & 7;
    v8f acc = {};
#pragma unroll
    for (int kt = 0; kt < 8; ++kt) {
      v16bf a = load_frag_A(sComb + kt * 32, 256, lane);
      v16bf w = load_frag_Bp(jiW + j * 256 * HID, nt * 8 + kt, lane);
      acc = wmma_bf16(a, w, acc);
    }
    int col = nt * 16 + (lane & 15);
    int rb  = (lane & 16) ? 8 : 0;
    float bias = jib[j * HID + col];
#pragma unroll
    for (int r = 0; r < 8; ++r)
      sX[(j * TILE_B + rb + r) * HID + col] = (bf16_t)fmaxf(acc[r] + bias, 0.0f);
  }
  for (int p = tid; p < TILE_B * HID; p += NTHREADS) sH[p] = (bf16_t)0.0f;
  float creg[8];
#pragma unroll
  for (int q = 0; q < 8; ++q) creg[q] = 0.0f;
  __syncthreads();

  // ---------------- time loop -----------------------------------------------
  for (int t = 0; t < TSTEPS; ++t) {
    // ===== GAT layers =====
    for (int l = 0; l < LAYERS; ++l) {
      const bf16_t* Wl = gatW + l * HID * HID;
      __builtin_prefetch(Wl + lane * 64, 0, 1);
      // xp = x @ W  (24 tiles of 16x16, K=128, nKt=4)
      for (int ti = wave; ti < NJ * 8; ti += NWAVES) {
        int j = ti >> 3, nt = ti & 7;
        v8f acc = {};
#pragma unroll
        for (int kt = 0; kt < 4; ++kt) {
          v16bf a = load_frag_A(sX + (j * TILE_B) * HID + kt * 32, HID, lane);
          v16bf w = load_frag_Bp(Wl, nt * 4 + kt, lane);
          acc = wmma_bf16(a, w, acc);
        }
        int col = nt * 16 + (lane & 15);
        int rb  = (lane & 16) ? 8 : 0;
#pragma unroll
        for (int r = 0; r < 8; ++r)
          sXP[(j * TILE_B + rb + r) * HID + col] = (bf16_t)acc[r];
      }
      __syncthreads();
      // attention scores s,d : [b][j][h]
      for (int p = tid; p < TILE_B * NJ * HEADS; p += NTHREADS) {
        int b = p / (NJ * HEADS), rem = p % (NJ * HEADS);
        int j = rem / HEADS, h = rem % HEADS;
        float ss = 0.0f, dd = 0.0f;
        const bf16_t* xp = sXP + (j * TILE_B + b) * HID + h * DH;
        for (int k = 0; k < DH; ++k) {
          float v = (float)xp[k];
          ss += v * gat_asrc[(l * HEADS + h) * DH + k];
          dd += v * gat_adst[(l * HEADS + h) * DH + k];
        }
        sS[p] = ss; sD[p] = dd;
      }
      __syncthreads();
      // masked leaky-relu softmax over j
      for (int p = tid; p < TILE_B * NJ * HEADS; p += NTHREADS) {
        int b = p / (NJ * HEADS), rem = p % (NJ * HEADS);
        int i = rem / HEADS, h = rem % HEADS;
        float e[NJ], m = -1e30f;
#pragma unroll
        for (int j = 0; j < NJ; ++j) {
          if (adj[i][j]) {
            float v = sD[(b * NJ + i) * HEADS + h] + sS[(b * NJ + j) * HEADS + h];
            v = (v > 0.0f) ? v : NEG_SLOPE * v;
            e[j] = v; if (v > m) m = v;
          } else e[j] = -1e30f;
        }
        float den = 0.0f, w_[NJ];
#pragma unroll
        for (int j = 0; j < NJ; ++j) {
          float x = (e[j] > -1e29f) ? __expf(e[j] - m) : 0.0f;
          w_[j] = x; den += x;
        }
#pragma unroll
        for (int j = 0; j < NJ; ++j)
          sAl[((b * NJ + i) * NJ + j) * HEADS + h] = w_[j] / den;
      }
      __syncthreads();
      // aggregate: out[i] = sum_j alpha[i,j,h] * xp[j] + gat_b
      for (int p = tid; p < NJ * TILE_B * HID; p += NTHREADS) {
        int f = p & 127, b = (p >> 7) & 15, i = p >> 11;
        int h = f >> 5;
        float acc = gat_b[l * HID + f];
#pragma unroll
        for (int j = 0; j < NJ; ++j)
          acc += sAl[((b * NJ + i) * NJ + j) * HEADS + h] *
                 (float)sXP[(j * TILE_B + b) * HID + f];
        sAgg[(i * TILE_B + b) * HID + f] = (bf16_t)acc;
      }
      __syncthreads();
      // layernorm + relu + residual (one thread per (joint,row))
      if (tid < NJ * TILE_B) {
        int i = tid / TILE_B, b = tid % TILE_B;
        const bf16_t* row = sAgg + (i * TILE_B + b) * HID;
        float mu = 0.0f;
        for (int f = 0; f < HID; ++f) mu += (float)row[f];
        mu *= (1.0f / HID);
        float var = 0.0f;
        for (int f = 0; f < HID; ++f) { float d0 = (float)row[f] - mu; var += d0 * d0; }
        var *= (1.0f / HID);
        float rstd = rsqrtf(var + 1e-5f);
        bf16_t* xrow = sX + (i * TILE_B + b) * HID;
        for (int f = 0; f < HID; ++f) {
          float xn = ((float)row[f] - mu) * rstd * ln_g[l * HID + f] + ln_b[l * HID + f];
          xrow[f] = (bf16_t)(fmaxf(xn, 0.0f) + (float)xrow[f]);
        }
      }
      __syncthreads();
    }

    // ===== LSTM gates: [16,128]@[128,512] x2, WMMA, 32 N-tiles (nKt=4) =====
    for (int nt = wave; nt < 32; nt += NWAVES) {
      v8f acc = {};
#pragma unroll
      for (int kt = 0; kt < 4; ++kt) {
        v16bf a = load_frag_A(sX + 2 * TILE_B * HID + kt * 32, HID, lane);
        v16bf w = load_frag_Bp(W_ih, nt * 4 + kt, lane);
        acc = wmma_bf16(a, w, acc);
      }
#pragma unroll
      for (int kt = 0; kt < 4; ++kt) {
        v16bf a = load_frag_A(sH + kt * 32, HID, lane);
        v16bf w = load_frag_Bp(W_hh, nt * 4 + kt, lane);
        acc = wmma_bf16(a, w, acc);
      }
      int col = nt * 16 + (lane & 15);
      int rb  = (lane & 16) ? 8 : 0;
      float bias = b_ih[col] + b_hh[col];
#pragma unroll
      for (int r = 0; r < 8; ++r)
        sG[(rb + r) * 512 + col] = (bf16_t)(acc[r] + bias);
    }
    __syncthreads();
    // elementwise cell update; c lives in registers
#pragma unroll
    for (int q = 0; q < 8; ++q) {
      int p = tid + q * NTHREADS;           // [0, 2048)
      int b = p >> 7, u = p & 127;
      float gi = (float)sG[b * 512 + u];
      float gf = (float)sG[b * 512 + 128 + u];
      float gg = (float)sG[b * 512 + 256 + u];
      float go = (float)sG[b * 512 + 384 + u];
      float cn = sigm(gf) * creg[q] + sigm(gi) * tanhf(gg);
      creg[q]  = cn;
      sH[p]    = (bf16_t)(sigm(go) * tanhf(cn));
    }
    __syncthreads();

    // ===== heads: p1 = relu(h@pW1+pb1), a1 = relu(h@aW1+ab1)  (nKt=4) =====
    {
      int nt = wave & 3;
      const bf16_t* Wm   = (wave < 4) ? pW1 : aW1;
      const float*  bias = (wave < 4) ? pb1 : ab1;
      bf16_t*       dst  = (wave < 4) ? sP1 : sA1;
      v8f acc = {};
#pragma unroll
      for (int kt = 0; kt < 4; ++kt) {
        v16bf a = load_frag_A(sH + kt * 32, HID, lane);
        v16bf w = load_frag_Bp(Wm, nt * 4 + kt, lane);
        acc = wmma_bf16(a, w, acc);
      }
      int col = nt * 16 + (lane & 15);
      int rb  = (lane & 16) ? 8 : 0;
      float bb = bias[col];
#pragma unroll
      for (int r = 0; r < 8; ++r)
        dst[(rb + r) * 64 + col] = (bf16_t)fmaxf(acc[r] + bb, 0.0f);
    }
    __syncthreads();
    // a2 = relu(a1@aW2+ab2)  (2 tiles, K=64, nKt=2)
    if (wave < 2) {
      int nt = wave;
      v8f acc = {};
#pragma unroll
      for (int kt = 0; kt < 2; ++kt) {
        v16bf a = load_frag_A(sA1 + kt * 32, 64, lane);
        v16bf w = load_frag_Bp(aW2, nt * 2 + kt, lane);
        acc = wmma_bf16(a, w, acc);
      }
      int col = nt * 16 + (lane & 15);
      int rb  = (lane & 16) ? 8 : 0;
      float bb = ab2[col];
#pragma unroll
      for (int r = 0; r < 8; ++r)
        sA2[(rb + r) * 32 + col] = (bf16_t)fmaxf(acc[r] + bb, 0.0f);
    }
    __syncthreads();
    // final 3-wide projections + combine, write out[b][t][3]
    if (tid < TILE_B * 3) {
      int b = tid / 3, cc = tid % 3;
      float phys = pb2[cc];
      for (int k = 0; k < 64; ++k) phys += (float)sP1[b * 64 + k] * pW2[k * 3 + cc];
      float av = ab3[cc];
      for (int k = 0; k < 32; ++k) av += (float)sA2[b * 32 + k] * aW3[k * 3 + cc];
      av += PHYS_W * phys;
      out[((gb + b) * TSTEPS + t) * 3 + cc] = av;
    }
    // noise carry for next step: x += 0.05*(t/T)*n
    float scale = 0.05f * ((float)t / (float)TSTEPS);
    for (int p = tid; p < NJ * TILE_B * HID; p += NTHREADS) {
      float x = (float)sX[p] + scale * hash_noise((unsigned)(blockIdx.x * 6144 + p),
                                                  (unsigned)t);
      sX[p] = (bf16_t)x;
    }
    __syncthreads();
  }
}

// ---------------------------------------------------------------------------
extern "C" void kernel_launch(void* const* d_in, const int* in_sizes, int n_in,
                              void* d_out, int out_size, void* d_ws, size_t ws_size,
                              hipStream_t stream) {
  (void)in_sizes; (void)n_in; (void)out_size; (void)ws_size;
  const int*   gesture_idx  = (const int*)  d_in[0];
  const float* demographics = (const float*)d_in[1];
  // d_in[2] = sequence_length (64, hardcoded)
  const float* emb_table = (const float*)d_in[3];
  const float* dW1  = (const float*)d_in[4];
  const float* db1  = (const float*)d_in[5];
  const float* dW2  = (const float*)d_in[6];
  const float* db2  = (const float*)d_in[7];
  const float* jiWf = (const float*)d_in[8];
  const float* jib  = (const float*)d_in[9];
  const float* gatWf    = (const float*)d_in[10];
  const float* gat_asrc = (const float*)d_in[11];
  const float* gat_adst = (const float*)d_in[12];
  const float* gat_b    = (const float*)d_in[13];
  const float* ln_g = (const float*)d_in[14];
  const float* ln_b = (const float*)d_in[15];
  const float* Wihf = (const float*)d_in[16];
  const float* Whhf = (const float*)d_in[17];
  const float* b_ih = (const float*)d_in[18];
  const float* b_hh = (const float*)d_in[19];
  const float* pW1f = (const float*)d_in[20];
  const float* pb1  = (const float*)d_in[21];
  const float* pW2  = (const float*)d_in[22];
  const float* pb2  = (const float*)d_in[23];
  const float* aW1f = (const float*)d_in[24];
  const float* ab1  = (const float*)d_in[25];
  const float* aW2f = (const float*)d_in[26];
  const float* ab2  = (const float*)d_in[27];
  const float* aW3  = (const float*)d_in[28];
  const float* ab3  = (const float*)d_in[29];

  bf16_t* ws = (bf16_t*)d_ws;
  auto repack = [&](const float* s, bf16_t* d, int K, int N, int slices) {
    int n = K * N * slices;
    repack_b_frag<<<(n + 255) / 256, 256, 0, stream>>>(s, d, K, N, slices);
  };
  repack(jiWf,  ws + OFF_JIW,  256, 128, NJ);
  repack(gatWf, ws + OFF_GATW, 128, 128, LAYERS);
  repack(Wihf,  ws + OFF_WIH,  128, 512, 1);
  repack(Whhf,  ws + OFF_WHH,  128, 512, 1);
  repack(pW1f,  ws + OFF_PW1,  128, 64, 1);
  repack(aW1f,  ws + OFF_AW1,  128, 64, 1);
  repack(aW2f,  ws + OFF_AW2,  64, 32, 1);

  vkc_main<<<NBLOCKS, NTHREADS, 0, stream>>>(
      gesture_idx, demographics, emb_table, dW1, db1, dW2, db2,
      ws + OFF_JIW, jib, ws + OFF_GATW, gat_asrc, gat_adst, gat_b, ln_g, ln_b,
      ws + OFF_WIH, ws + OFF_WHH, b_ih, b_hh,
      ws + OFF_PW1, pb1, pW2, pb2,
      ws + OFF_AW1, ab1, ws + OFF_AW2, ab2, aW3, ab3,
      (float*)d_out);
}
